// DrugEncoder_27066883899917
// MI455X (gfx1250) — matrix operand order repaired and checked by reading, not compiled, not run on hardware
//
#include <hip/hip_runtime.h>
#include <stdint.h>

#define HID 128
#define WT_STRIDE 144   // 128 + 16 halves pad -> 288B row stride (32B-aligned v16h loads)

typedef __attribute__((ext_vector_type(16))) _Float16 v16h;
typedef __attribute__((ext_vector_type(8)))  float    v8f;

// ---------------- init: dinv=1 (self loop), counts=0, out=0 ----------------
__global__ void dk_init(float* dinv, float* counts, float* out,
                        int n, int g, int outtotal) {
  int t = blockIdx.x * blockDim.x + threadIdx.x;
  if (t < n) dinv[t] = 1.0f;
  if (t < g) counts[t] = 0.0f;
  if (t < outtotal) out[t] = 0.0f;
}

// ---------------- degree accumulation on dst side ----------------
__global__ void dk_deg_accum(const int* __restrict__ dst, float* deg, int e) {
  int t = blockIdx.x * blockDim.x + threadIdx.x;
  if (t >= e) return;
  atomicAdd(&deg[dst[t]], 1.0f);
}

__global__ void dk_deg_fin(float* dinv, int n) {
  int t = blockIdx.x * blockDim.x + threadIdx.x;
  if (t < n) dinv[t] = rsqrtf(dinv[t]);
}

// ---------------- WMMA GEMM: Hout[M,128] = act(X[M,128]) @ W[128,128] ------
// Wave computes 16 rows x 128 cols. A: 16x32 f16 frags, contiguous k = 16*half + j,
// B from LDS-transposed f16 weight panel. 32 v_wmma per wave.
template <bool LEAKY>
__global__ __launch_bounds__(256) void dk_gemm(const float* __restrict__ X,
                                               const float* __restrict__ W,
                                               float* __restrict__ Hout,
                                               int nrows) {
  __shared__ __align__(32) _Float16 WT[HID * WT_STRIDE];

  // Cooperative load: WT[n][k] = (f16) W[k][n]   (coalesced global reads)
  for (int idx = threadIdx.x; idx < HID * HID; idx += 256) {
    int k = idx >> 7;
    int n = idx & 127;
    WT[n * WT_STRIDE + k] = (_Float16)W[idx];
  }
  __syncthreads();

  const int wave = threadIdx.x >> 5;
  const int lane = threadIdx.x & 31;
  const int rowbase = (blockIdx.x * 8 + wave) * 16;
  if (rowbase >= nrows) return;

  const int half = lane >> 4;          // K-group selector: k = 16*half + j (contiguous)
  const int l16  = lane & 15;
  const float* xrow = X + (size_t)(rowbase + l16) * HID;

  v8f acc[8] = {};

#pragma unroll
  for (int kc = 0; kc < 4; kc++) {
    // ---- A fragment: halves 0..15 -> K = kc*32 + half*16 + j
    v16h a;
    {
      const int k0 = kc * 32 + half * 16;
      float4 q0 = *(const float4*)(xrow + k0);
      float4 q1 = *(const float4*)(xrow + k0 + 4);
      float4 q2 = *(const float4*)(xrow + k0 + 8);
      float4 q3 = *(const float4*)(xrow + k0 + 12);
      float f[16] = {q0.x, q0.y, q0.z, q0.w, q1.x, q1.y, q1.z, q1.w,
                     q2.x, q2.y, q2.z, q2.w, q3.x, q3.y, q3.z, q3.w};
#pragma unroll
      for (int j = 0; j < 16; j++) {
        float v = f[j];
        if (LEAKY) v = v > 0.0f ? v : 0.01f * v;
        a[j] = (_Float16)v;
      }
    }
    // ---- 8 N-tiles: B frag = column (nt*16 + l16), K = kc*32 + half*16 + j
#pragma unroll
    for (int nt = 0; nt < 8; nt++) {
      const v16h b = *(const v16h*)(&WT[(nt * 16 + l16) * WT_STRIDE + kc * 32 + half * 16]);
      acc[nt] = __builtin_amdgcn_wmma_f32_16x16x32_f16(
          false, a, false, b, (short)0, acc[nt], false, false);
    }
  }

  // ---- store C: vgpr r -> row = rowbase + r + 8*half, col = nt*16 + l16
#pragma unroll
  for (int nt = 0; nt < 8; nt++) {
    const int col = nt * 16 + l16;
#pragma unroll
    for (int r = 0; r < 8; r++) {
      const int row = rowbase + r + half * 8;
      Hout[(size_t)row * HID + col] = acc[nt][r];
    }
  }
}

// ---------------- AGG init: self-loop term + bias ----------------
__global__ void dk_agg_init(const float* __restrict__ H, const float* __restrict__ dinv,
                            const float* __restrict__ bias, float* __restrict__ AGG,
                            int total) {
  int t = blockIdx.x * blockDim.x + threadIdx.x;
  if (t >= total) return;
  int i = t >> 7;
  int d = t & 127;
  float di = dinv[i];
  AGG[t] = H[t] * di * di + bias[d];
}

// ---------------- edge scatter: one wave per edge ----------------
__global__ void dk_edge_scatter(const float* __restrict__ H, const float* __restrict__ dinv,
                                const int* __restrict__ src, const int* __restrict__ dst,
                                float* __restrict__ AGG, int e) {
  int t = blockIdx.x * blockDim.x + threadIdx.x;
  int ed = t >> 5;
  int lane = t & 31;
  if (ed >= e) return;
  int s = src[ed];
  int d = dst[ed];
  float c = dinv[s] * dinv[d];
  float4 v = *(const float4*)(H + (size_t)s * HID + lane * 4);
  float* o = AGG + (size_t)d * HID + lane * 4;
  atomicAdd(o + 0, v.x * c);
  atomicAdd(o + 1, v.y * c);
  atomicAdd(o + 2, v.z * c);
  atomicAdd(o + 3, v.w * c);
}

// ---------------- pooling: leaky-relu fused, wave per node ----------------
__device__ __forceinline__ float dk_leaky(float x) { return x > 0.0f ? x : 0.01f * x; }

__global__ void dk_pool_scatter(const float* __restrict__ AGG, const int* __restrict__ batch,
                                float* __restrict__ outsum, float* __restrict__ counts, int n) {
  int t = blockIdx.x * blockDim.x + threadIdx.x;
  int i = t >> 5;
  int lane = t & 31;
  if (i >= n) return;
  int g = batch[i];
  if (lane == 0) atomicAdd(&counts[g], 1.0f);
  float4 v = *(const float4*)(AGG + (size_t)i * HID + lane * 4);
  float* o = outsum + (size_t)g * HID + lane * 4;
  atomicAdd(o + 0, dk_leaky(v.x));
  atomicAdd(o + 1, dk_leaky(v.y));
  atomicAdd(o + 2, dk_leaky(v.z));
  atomicAdd(o + 3, dk_leaky(v.w));
}

__global__ void dk_pool_div(float* out, const float* __restrict__ counts, int total) {
  int t = blockIdx.x * blockDim.x + threadIdx.x;
  if (t >= total) return;
  int g = t >> 7;
  out[t] /= fmaxf(counts[g], 1.0f);
}

// ======================================================================
extern "C" void kernel_launch(void* const* d_in, const int* in_sizes, int n_in,
                              void* d_out, int out_size, void* d_ws, size_t ws_size,
                              hipStream_t stream) {
  const float* X  = (const float*)d_in[0];
  const float* W1 = (const float*)d_in[1];
  const float* b1 = (const float*)d_in[2];
  const float* W2 = (const float*)d_in[3];
  const float* b2 = (const float*)d_in[4];
  const int*   ei = (const int*)d_in[5];
  const int* batch = (const int*)d_in[6];
  float* out = (float*)d_out;

  const int N = in_sizes[0] / HID;   // 50000 nodes
  const int E = in_sizes[5] / 2;     // 600000 edges
  const int G = out_size / HID;      // 1000 graphs

  const int* src = ei;
  const int* dst = ei + E;

  // workspace layout
  float* dinv   = (float*)d_ws;
  float* counts = dinv + N;
  uintptr_t p = (uintptr_t)(counts + G);
  p = (p + 255) & ~(uintptr_t)255;
  float* bufA = (float*)p;                 // H1 / H2
  float* bufB = bufA + (size_t)N * HID;    // AGG1 / AGG2

  const int TB = 256;
  const int totNH = N * HID;
  const int outTot = G * HID;
  const int initN = (N > outTot) ? N : outTot;

  // 0) init
  dk_init<<<(initN + TB - 1) / TB, TB, 0, stream>>>(dinv, counts, out, N, G, outTot);
  // 1) degree (deg in dinv), then rsqrt
  dk_deg_accum<<<(E + TB - 1) / TB, TB, 0, stream>>>(dst, dinv, E);
  dk_deg_fin<<<(N + TB - 1) / TB, TB, 0, stream>>>(dinv, N);

  const int rows16 = (N + 15) / 16;
  const int gemmBlocks = (rows16 + 7) / 8;

  // 2) layer 1: H1 = X @ W1 (WMMA)
  dk_gemm<false><<<gemmBlocks, TB, 0, stream>>>(X, W1, bufA, N);
  // 3) AGG1 = H1*dinv^2 + b1, then edge scatter
  dk_agg_init<<<(totNH + TB - 1) / TB, TB, 0, stream>>>(bufA, dinv, b1, bufB, totNH);
  dk_edge_scatter<<<((E * 32) + TB - 1) / TB, TB, 0, stream>>>(bufA, dinv, src, dst, bufB, E);

  // 4) layer 2: H2 = leaky(AGG1) @ W2 (leaky fused into A load)
  dk_gemm<true><<<gemmBlocks, TB, 0, stream>>>(bufB, W2, bufA, N);
  // 5) AGG2 = H2*dinv^2 + b2, then edge scatter
  dk_agg_init<<<(totNH + TB - 1) / TB, TB, 0, stream>>>(bufA, dinv, b2, bufB, totNH);
  dk_edge_scatter<<<((E * 32) + TB - 1) / TB, TB, 0, stream>>>(bufA, dinv, src, dst, bufB, E);

  // 6) global mean pool (leaky fused)
  dk_pool_scatter<<<((N * 32) + TB - 1) / TB, TB, 0, stream>>>(bufB, batch, out, counts, N);
  dk_pool_div<<<(outTot + TB - 1) / TB, TB, 0, stream>>>(out, counts, outTot);
}
